// LocalWindowAttention_89790586290381
// MI455X (gfx1250) — matrix operand order repaired and checked by reading, not compiled, not run on hardware
//
#include <hip/hip_runtime.h>
#include <hip/hip_bf16.h>
#include <math.h>
#include <stdint.h>

// ---------------------------------------------------------------------------
// LocalWindowAttention transformer layer for MI455X (gfx1250), bf16 WMMA path.
// B=8 T=2048 D=1024 W=16 H=8 HD=128.  All matmuls via v_wmma_f32_16x16x32_bf16.
// LDS staging via GLOBAL_LOAD_ASYNC_TO_LDS (ASYNCcnt) — CDNA5 async path.
// ---------------------------------------------------------------------------

typedef __bf16 v16bf __attribute__((ext_vector_type(16)));
typedef float  v8f   __attribute__((ext_vector_type(8)));

#define NTOK   16384          // B*T
#define DMODEL 1024
#define DQKV   3072
#define DFFN   4096
#define NEG_LOG1E4_OVER_D (-0.0089944730195081f)   // -ln(10000)/1024

// ---------------- WMMA fragment helpers (ISA 7.12.2 layouts) ----------------

__device__ __forceinline__ v8f wmma_bf(v16bf a, v16bf b, v8f c) {
  return __builtin_amdgcn_wmma_f32_16x16x32_bf16(false, a, false, b,
                                                 (short)0, c, false, false);
}

// Async 16B copy global -> LDS (ISA §10.7 GLOBAL_LOAD_ASYNC_TO_LDS_B128).
// LDS operand = low 32 bits of the flat shared pointer (LDS aperture offset).
__device__ __forceinline__ void async_cp16(void* lds, const void* g) {
  unsigned loff = (unsigned)(uintptr_t)lds;
  unsigned long long ga = (unsigned long long)(uintptr_t)g;
  asm volatile("global_load_async_to_lds_b128 %0, %1, off"
               :: "v"(loff), "v"(ga) : "memory");
}
__device__ __forceinline__ void wait_async0() {
  asm volatile("s_wait_asynccnt 0" ::: "memory");
}

// A-matrix 16x32 bf16 fragment from row-major LDS source at (0, k0)
// lane L: M = L&15 ; element e -> K = (e&7) + 16*(e>>3) + 8*(L>>4)
__device__ __forceinline__ v16bf build_a(const __bf16* s, int ld, int lane) {
  const int M  = lane & 15;
  const int kh = (lane >> 4) << 3;
  const __bf16* p = s + M * ld + kh;
  union { v16bf v; uint2 q[4]; } f;
  f.q[0] = *(const uint2*)(p);
  f.q[1] = *(const uint2*)(p + 4);
  f.q[2] = *(const uint2*)(p + 16);
  f.q[3] = *(const uint2*)(p + 20);
  return f.v;
}

// Same but only K=0..15 valid (upper half zero) — used for P in attention
__device__ __forceinline__ v16bf build_a_k16(const __bf16* s, int ld, int lane) {
  const int M  = lane & 15;
  const int kh = (lane >> 4) << 3;
  const __bf16* p = s + M * ld + kh;
  union { v16bf v; uint2 q[4]; } f;
  f.q[0] = *(const uint2*)(p);
  f.q[1] = *(const uint2*)(p + 4);
  f.q[2] = make_uint2(0u, 0u);
  f.q[3] = make_uint2(0u, 0u);
  return f.v;
}

// B-matrix 32x16 bf16 fragment; source is "weight layout" w[n][k] row-major:
// lane L: N = L&15 ; element e -> K = (e&7) + 16*(e>>3) + 8*(L>>4)
__device__ __forceinline__ v16bf build_b(const __bf16* w, long ld, int lane) {
  const int n  = lane & 15;
  const int kh = (lane >> 4) << 3;
  const __bf16* p = w + (size_t)n * ld + kh;
  union { v16bf v; uint4 q[2]; } f;
  f.q[0] = *(const uint4*)(p);
  f.q[1] = *(const uint4*)(p + 16);
  return f.v;
}

// B fragment with only K=0..15 valid, source stored K-major s[k][n] (e.g. V)
__device__ __forceinline__ v16bf build_b_k16_cm(const __bf16* s, int ld, int lane) {
  union { v16bf v; unsigned u[8]; } f;
  {
    const __bf16* p = s + (lane & 15) + ((lane >> 4) << 3) * ld;
#pragma unroll
    for (int e = 0; e < 8; ++e) f.v[e] = p[e * ld];
#pragma unroll
    for (int i = 4; i < 8; ++i) f.u[i] = 0u;
  }
  return f.v;
}

__device__ __forceinline__ float pe_val(float pos, int col) {
  float freq = __expf(NEG_LOG1E4_OVER_D * (float)(col & ~1));
  float ang  = pos * freq;
  return (col & 1) ? __cosf(ang) : __sinf(ang);
}

// ------------------------------ K0: f32 -> bf16 -----------------------------
__global__ __launch_bounds__(256) void k_cvt(const float* __restrict__ s,
                                             __bf16* __restrict__ d, int n) {
  int i = blockIdx.x * 256 + threadIdx.x;
  if (i < n) d[i] = (__bf16)s[i];
}

// ------------------------- K1: x = features + PE ---------------------------
__global__ __launch_bounds__(256) void k_prep(const float* __restrict__ feat,
                                              const int* __restrict__ fidx,
                                              __bf16* __restrict__ xb) {
  size_t e = (size_t)blockIdx.x * 256 + threadIdx.x;   // < 16384*1024
  int token = (int)(e >> 10);
  int col   = (int)(e & 1023);
  float pos = (float)fidx[token];
  xb[e] = (__bf16)(feat[e] + pe_val(pos, col));
}

// ---------------- K2: QKV GEMM  qkv[16384][3072] = xb * Wqkv^T + b ----------
// grid (24 Nblocks, 256 Mblocks), 256 thr / 8 waves, BM=64, BN=128
__global__ __launch_bounds__(256) void k_gemm_qkv(const __bf16* __restrict__ xb,
                                                  const __bf16* __restrict__ wq,
                                                  const float* __restrict__ bq,
                                                  __bf16* __restrict__ qkv) {
  extern __shared__ char smem[];
  __bf16* sA = (__bf16*)smem;                     // [64][1032]
  const int mb = blockIdx.y, nb = blockIdx.x;
  const int tid = threadIdx.x, wave = tid >> 5, lane = tid & 31;

  {   // async-stage A block (64 rows x 1024) into LDS, 16B chunks
    const __bf16* src = xb + (size_t)mb * 64 * DMODEL;
    for (int c = tid; c < 8192; c += 256) {
      int row = c >> 7, col8 = c & 127;
      async_cp16(sA + row * 1032 + col8 * 8, src + (size_t)c * 8);
    }
  }
  wait_async0();
  __syncthreads();

  const int ntile = nb * 8 + wave;                // 0..191
  const int n0 = ntile * 16;
  v8f acc[4] = {};
  const __bf16* wrow = wq + (size_t)n0 * DMODEL;
#pragma unroll 2
  for (int k0 = 0; k0 < DMODEL; k0 += 32) {
    __builtin_prefetch(wrow + k0 + 32, 0, 3);     // warm next k-step B lines
    v16bf bf = build_b(wrow + k0, DMODEL, lane);
    v16bf a0 = build_a(sA + k0, 1032, lane);
    v16bf a1 = build_a(sA + 16 * 1032 + k0, 1032, lane);
    v16bf a2 = build_a(sA + 32 * 1032 + k0, 1032, lane);
    v16bf a3 = build_a(sA + 48 * 1032 + k0, 1032, lane);
    acc[0] = wmma_bf(a0, bf, acc[0]);
    acc[1] = wmma_bf(a1, bf, acc[1]);
    acc[2] = wmma_bf(a2, bf, acc[2]);
    acc[3] = wmma_bf(a3, bf, acc[3]);
  }
  const int col   = n0 + (lane & 15);
  const int rbase = (lane >> 4) << 3;
  const float bias = bq[col];
#pragma unroll
  for (int mf = 0; mf < 4; ++mf)
#pragma unroll
    for (int v = 0; v < 8; ++v) {
      int row = mb * 64 + mf * 16 + rbase + v;
      qkv[(size_t)row * DQKV + col] = (__bf16)(acc[mf][v] + bias);
    }
}

// ------------- K3: windowed attention, 1 WG / window, 1 wave / head --------
__global__ __launch_bounds__(256) void k_attn(const __bf16* __restrict__ qkv,
                                              __bf16* __restrict__ ao) {
  extern __shared__ char smem[];
  __bf16* sQ = (__bf16*)smem;            // 8 x [16][136]
  __bf16* sK = sQ + 8 * 2176;
  __bf16* sV = sK + 8 * 2176;
  __bf16* sP = sV + 8 * 2176;            // 8 x [16][24]
  const int win = blockIdx.x;            // 0..1023
  const int tid = threadIdx.x, h = tid >> 5, lane = tid & 31;
  __bf16* sQh = sQ + h * 2176;
  __bf16* sKh = sK + h * 2176;
  __bf16* sVh = sV + h * 2176;
  __bf16* sPh = sP + h * 384;

  const size_t base = (size_t)win * 16 * DQKV + h * 128;
  const int c8 = lane & 15;
#pragma unroll
  for (int i = 0; i < 8; ++i) {          // async-stage Q,K,V head slices
    int row = i * 2 + (lane >> 4);
    const __bf16* r = qkv + base + (size_t)row * DQKV;
    async_cp16(sQh + row * 136 + c8 * 8, r + c8 * 8);
    async_cp16(sKh + row * 136 + c8 * 8, r + 1024 + c8 * 8);
    async_cp16(sVh + row * 136 + c8 * 8, r + 2048 + c8 * 8);
  }
  wait_async0();                         // per-wave private tiles: no barrier

  // S = Q K^T (per-head 16x16, K-dim 128 -> 4 WMMAs)
  v8f s = {};
  for (int kc = 0; kc < 128; kc += 32) {
    v16bf qf = build_a(sQh + kc, 136, lane);
    v16bf kf = build_b(sKh + kc, 136, lane);   // B[d][ktok] = K[ktok][d]
    s = wmma_bf(qf, kf, s);
  }
  // softmax over columns: row M lives across a 16-lane group, fixed vgpr v
  const int rbase = (lane >> 4) << 3;
  const float scale = 0.08838834764831845f;    // 1/sqrt(128)
#pragma unroll
  for (int v = 0; v < 8; ++v) {
    float x = s[v] * scale;
    float m = x;
    for (int d = 1; d < 16; d <<= 1) m = fmaxf(m, __shfl_xor(m, d, 32));
    float e = __expf(x - m);
    float t = e;
    for (int d = 1; d < 16; d <<= 1) t += __shfl_xor(t, d, 32);
    float p = e / t;
    sPh[(v + rbase) * 24 + (lane & 15)] = (__bf16)p;   // P row-major
  }
  // O = P V : 8 N-tiles of 16, single WMMA each (K padded 16->32)
  v16bf pa = build_a_k16(sPh, 24, lane);
#pragma unroll
  for (int nt = 0; nt < 8; ++nt) {
    v16bf vb = build_b_k16_cm(sVh + nt * 16, 136, lane);
    v8f o = {};
    o = wmma_bf(pa, vb, o);
#pragma unroll
    for (int v = 0; v < 8; ++v)                      // repack into LDS (reuse sK)
      sKh[(v + rbase) * 136 + nt * 16 + (lane & 15)] = (__bf16)o[v];
  }
#pragma unroll
  for (int i = 0; i < 8; ++i) {                      // coalesced store of O
    int row = i * 2 + (lane >> 4);
    *(uint4*)(ao + (size_t)(win * 16 + row) * DMODEL + h * 128 + c8 * 8) =
        *(const uint4*)(sKh + row * 136 + c8 * 8);
  }
}

// ------- K4: out-proj GEMM + bias + residual(x=feat+PE) + LayerNorm1 -------
// grid 512 Mblocks of 32 rows; per-wave 8 N-tiles x 2 M-frags (128 f32 acc)
__global__ __launch_bounds__(256) void k_outproj_ln1(
    const __bf16* __restrict__ ao, const __bf16* __restrict__ wo,
    const float* __restrict__ bo, const float* __restrict__ feat,
    const int* __restrict__ fidx, const float* __restrict__ g1,
    const float* __restrict__ be1, __bf16* __restrict__ hb) {
  extern __shared__ char smem[];
  __bf16* sA = (__bf16*)smem;                       // [32][1032]
  __shared__ float rsum[32], rsq[32];
  const int mb = blockIdx.x;
  const int tid = threadIdx.x, wave = tid >> 5, lane = tid & 31;
  if (tid < 32) { rsum[tid] = 0.f; rsq[tid] = 0.f; }
  {
    const __bf16* src = ao + (size_t)mb * 32 * DMODEL;
    for (int c = tid; c < 4096; c += 256) {
      int row = c >> 7, col8 = c & 127;
      async_cp16(sA + row * 1032 + col8 * 8, src + (size_t)c * 8);
    }
  }
  wait_async0();
  __syncthreads();

  v8f acc[16] = {};                                  // [i=0..7][mf=0..1]
  for (int k0 = 0; k0 < DMODEL; k0 += 32) {
    v16bf af0 = build_a(sA + k0, 1032, lane);
    v16bf af1 = build_a(sA + 16 * 1032 + k0, 1032, lane);
    __builtin_prefetch(wo + (size_t)(wave * 8) * 16 * DMODEL + k0 + 32, 0, 3);
    v16bf bf = build_b(wo + (size_t)(wave * 8) * 16 * DMODEL + k0, DMODEL, lane);
#pragma unroll
    for (int i = 0; i < 8; ++i) {                    // double-buffered B frags
      v16bf bcur = bf;
      if (i < 7)
        bf = build_b(wo + (size_t)((wave * 8 + i + 1) * 16) * DMODEL + k0, DMODEL, lane);
      acc[i * 2 + 0] = wmma_bf(af0, bcur, acc[i * 2 + 0]);
      acc[i * 2 + 1] = wmma_bf(af1, bcur, acc[i * 2 + 1]);
    }
  }
  const int rbase = (lane >> 4) << 3;
  // compose: + bias + residual (features + PE recomputed)
#pragma unroll
  for (int i = 0; i < 8; ++i) {
    int col = (wave * 8 + i) * 16 + (lane & 15);
    float bv = bo[col];
#pragma unroll
    for (int mf = 0; mf < 2; ++mf)
#pragma unroll
      for (int v = 0; v < 8; ++v) {
        int row = mf * 16 + rbase + v;
        int token = mb * 32 + row;
        float f = acc[i * 2 + mf][v] + bv +
                  feat[(size_t)token * DMODEL + col] +
                  pe_val((float)fidx[token], col);
        acc[i * 2 + mf][v] = f;
      }
  }
  // row stats via shuffle + LDS f32 atomics
#pragma unroll
  for (int mf = 0; mf < 2; ++mf)
#pragma unroll
    for (int v = 0; v < 8; ++v) {
      float s1 = 0.f, s2 = 0.f;
#pragma unroll
      for (int i = 0; i < 8; ++i) { float f = acc[i * 2 + mf][v]; s1 += f; s2 += f * f; }
      for (int d = 1; d < 16; d <<= 1) { s1 += __shfl_xor(s1, d, 32); s2 += __shfl_xor(s2, d, 32); }
      if ((lane & 15) == 0) {
        int row = mf * 16 + rbase + v;
        atomicAdd(&rsum[row], s1);
        atomicAdd(&rsq[row], s2);
      }
    }
  __syncthreads();
#pragma unroll
  for (int mf = 0; mf < 2; ++mf)
#pragma unroll
    for (int v = 0; v < 8; ++v) {
      int row = mf * 16 + rbase + v;
      float mean = rsum[row] * (1.0f / 1024.0f);
      float var  = rsq[row] * (1.0f / 1024.0f) - mean * mean;
      float rstd = rsqrtf(var + 1e-5f);
      int token = mb * 32 + row;
#pragma unroll
      for (int i = 0; i < 8; ++i) {
        int col = (wave * 8 + i) * 16 + (lane & 15);
        float val = (acc[i * 2 + mf][v] - mean) * rstd * g1[col] + be1[col];
        hb[(size_t)token * DMODEL + col] = (__bf16)val;
      }
    }
}

// --------- K5: fused FFN (GEMM1+GELU+GEMM2, chunked hidden) + LN2 ----------
__global__ __launch_bounds__(256) void k_ffn_ln2(
    const __bf16* __restrict__ hb, const __bf16* __restrict__ w1b,
    const float* __restrict__ b1, const __bf16* __restrict__ w2b,
    const float* __restrict__ b2, const float* __restrict__ g2,
    const float* __restrict__ be2, float* __restrict__ out) {
  extern __shared__ char smem[];
  __bf16* sH = (__bf16*)smem;                       // [32][1032]
  __bf16* sU = sH + 32 * 1032;                      // [32][136]
  __shared__ float rsum[32], rsq[32];
  const int mb = blockIdx.x;
  const int tid = threadIdx.x, wave = tid >> 5, lane = tid & 31;
  if (tid < 32) { rsum[tid] = 0.f; rsq[tid] = 0.f; }
  {
    const __bf16* src = hb + (size_t)mb * 32 * DMODEL;
    for (int c = tid; c < 4096; c += 256) {
      int row = c >> 7, col8 = c & 127;
      async_cp16(sH + row * 1032 + col8 * 8, src + (size_t)c * 8);
    }
  }
  wait_async0();
  __syncthreads();

  const int rbase = (lane >> 4) << 3;
  v8f acc[16] = {};                                  // final out accumulators
  for (int c = 0; c < 32; ++c) {                     // hidden chunks of 128
    // FFN1: wave computes hidden tile (c*128 + wave*16 .. +15) for both M-frags
    v8f u0 = {}, u1 = {};
    const __bf16* w1row = w1b + (size_t)(c * 128 + wave * 16) * DMODEL;
#pragma unroll 2
    for (int k0 = 0; k0 < DMODEL; k0 += 32) {
      __builtin_prefetch(w1row + k0 + 32, 0, 3);
      v16bf bf = build_b(w1row + k0, DMODEL, lane);
      v16bf af0 = build_a(sH + k0, 1032, lane);
      v16bf af1 = build_a(sH + 16 * 1032 + k0, 1032, lane);
      u0 = wmma_bf(af0, bf, u0);
      u1 = wmma_bf(af1, bf, u1);
    }
    const int hcol = wave * 16 + (lane & 15);
    const float b1v = b1[c * 128 + hcol];
#pragma unroll
    for (int v = 0; v < 8; ++v) {                    // exact GELU -> LDS bf16
      float x = u0[v] + b1v;
      sU[(rbase + v) * 136 + hcol] = (__bf16)(0.5f * x * (1.f + erff(x * 0.70710678f)));
      float y = u1[v] + b1v;
      sU[(16 + rbase + v) * 136 + hcol] = (__bf16)(0.5f * y * (1.f + erff(y * 0.70710678f)));
    }
    __syncthreads();
    // FFN2: accumulate this chunk into the 8 output N-tiles per wave
    for (int k0 = 0; k0 < 128; k0 += 32) {
      v16bf af0 = build_a(sU + k0, 136, lane);
      v16bf af1 = build_a(sU + 16 * 136 + k0, 136, lane);
      v16bf bf = build_b(w2b + (size_t)(wave * 8) * 16 * DFFN + c * 128 + k0, DFFN, lane);
#pragma unroll
      for (int i = 0; i < 8; ++i) {                  // double-buffered B frags
        v16bf bcur = bf;
        if (i < 7)
          bf = build_b(w2b + (size_t)((wave * 8 + i + 1) * 16) * DFFN + c * 128 + k0,
                       DFFN, lane);
        acc[i * 2 + 0] = wmma_bf(af0, bcur, acc[i * 2 + 0]);
        acc[i * 2 + 1] = wmma_bf(af1, bcur, acc[i * 2 + 1]);
      }
    }
    __syncthreads();                                 // sU reused next chunk
  }
  // compose: + b2 + residual h, then LN2, write f32 output
#pragma unroll
  for (int i = 0; i < 8; ++i) {
    int col = (wave * 8 + i) * 16 + (lane & 15);
    float bv = b2[col];
#pragma unroll
    for (int mf = 0; mf < 2; ++mf)
#pragma unroll
      for (int v = 0; v < 8; ++v) {
        int row = mf * 16 + rbase + v;
        float f = acc[i * 2 + mf][v] + bv + (float)sH[row * 1032 + col];
        acc[i * 2 + mf][v] = f;
      }
  }
#pragma unroll
  for (int mf = 0; mf < 2; ++mf)
#pragma unroll
    for (int v = 0; v < 8; ++v) {
      float s1 = 0.f, s2 = 0.f;
#pragma unroll
      for (int i = 0; i < 8; ++i) { float f = acc[i * 2 + mf][v]; s1 += f; s2 += f * f; }
      for (int d = 1; d < 16; d <<= 1) { s1 += __shfl_xor(s1, d, 32); s2 += __shfl_xor(s2, d, 32); }
      if ((lane & 15) == 0) {
        int row = mf * 16 + rbase + v;
        atomicAdd(&rsum[row], s1);
        atomicAdd(&rsq[row], s2);
      }
    }
  __syncthreads();
#pragma unroll
  for (int mf = 0; mf < 2; ++mf)
#pragma unroll
    for (int v = 0; v < 8; ++v) {
      int row = mf * 16 + rbase + v;
      float mean = rsum[row] * (1.0f / 1024.0f);
      float var  = rsq[row] * (1.0f / 1024.0f) - mean * mean;
      float rstd = rsqrtf(var + 1e-5f);
      int token = mb * 32 + row;
#pragma unroll
      for (int i = 0; i < 8; ++i) {
        int col = (wave * 8 + i) * 16 + (lane & 15);
        out[(size_t)token * DMODEL + col] =
            (acc[i * 2 + mf][v] - mean) * rstd * g2[col] + be2[col];
      }
    }
}

// ---------------------------------------------------------------------------
extern "C" void kernel_launch(void* const* d_in, const int* in_sizes, int n_in,
                              void* d_out, int out_size, void* d_ws, size_t ws_size,
                              hipStream_t stream) {
  const float* feat  = (const float*)d_in[0];
  const int*   fidx  = (const int*)d_in[1];
  const float* w_qkv = (const float*)d_in[2];
  const float* b_qkv = (const float*)d_in[3];
  const float* w_out = (const float*)d_in[4];
  const float* b_out = (const float*)d_in[5];
  const float* w1    = (const float*)d_in[6];
  const float* b1    = (const float*)d_in[7];
  const float* w2    = (const float*)d_in[8];
  const float* b2    = (const float*)d_in[9];
  const float* g1    = (const float*)d_in[10];
  const float* be1   = (const float*)d_in[11];
  const float* g2    = (const float*)d_in[12];
  const float* be2   = (const float*)d_in[13];
  float* out = (float*)d_out;

  // workspace layout (regions aliased; ~152 MB total)
  char* p = (char*)d_ws;
  __bf16* xb_ao  = (__bf16*)p; p += (size_t)NTOK * DMODEL * 2;   // xb (K1/K2) then ao (K3/K4)
  __bf16* qkv_hb = (__bf16*)p; p += (size_t)NTOK * DQKV * 2;     // qkv (K2/K3) then hb (K4/K5)
  __bf16* wqb    = (__bf16*)p; p += (size_t)DQKV * DMODEL * 2;
  __bf16* wob    = (__bf16*)p; p += (size_t)DMODEL * DMODEL * 2;
  __bf16* w1b    = (__bf16*)p; p += (size_t)DFFN * DMODEL * 2;
  __bf16* w2b    = (__bf16*)p; p += (size_t)DMODEL * DFFN * 2;

  k_cvt<<<dim3(DQKV * DMODEL / 256), 256, 0, stream>>>(w_qkv, wqb, DQKV * DMODEL);
  k_cvt<<<dim3(DMODEL * DMODEL / 256), 256, 0, stream>>>(w_out, wob, DMODEL * DMODEL);
  k_cvt<<<dim3(DFFN * DMODEL / 256), 256, 0, stream>>>(w1, w1b, DFFN * DMODEL);
  k_cvt<<<dim3(DMODEL * DFFN / 256), 256, 0, stream>>>(w2, w2b, DMODEL * DFFN);

  k_prep<<<dim3(NTOK * DMODEL / 256), 256, 0, stream>>>(feat, fidx, xb_ao);

  k_gemm_qkv<<<dim3(24, 256), 256, 64 * 1032 * 2, stream>>>(xb_ao, wqb, b_qkv, qkv_hb);

  k_attn<<<dim3(NTOK / 16), 256, (3 * 8 * 2176 + 8 * 384) * 2, stream>>>(qkv_hb, xb_ao);

  k_outproj_ln1<<<dim3(NTOK / 32), 256, 32 * 1032 * 2, stream>>>(
      xb_ao, wob, b_out, feat, fidx, g1, be1, qkv_hb);

  k_ffn_ln2<<<dim3(NTOK / 32), 256, (32 * 1032 + 32 * 136) * 2, stream>>>(
      qkv_hb, w1b, b1, w2b, b2, g2, be2, out);
}
